// GraphEncoder_46033459479126
// MI455X (gfx1250) — matrix operand order repaired and checked by reading, not compiled, not run on hardware
//
#include <hip/hip_runtime.h>
#include <hip/hip_bf16.h>

#define N_NODES 20000
#define N_EDGES 128000
#define N_GRAPHS 64
#define N_EMB 10000
#define N_TYPES 100
#define FEAT 100
#define FEATP 128          // FEAT padded to K multiple of 32
#define HID 256
#define HEADS 4
#define HC 1024

typedef __attribute__((ext_vector_type(16))) __bf16 v16bf;
typedef __attribute__((ext_vector_type(8)))  float  v8f;
typedef __attribute__((ext_vector_type(4)))  unsigned int u32x4;
typedef __attribute__((ext_vector_type(8)))  int i32x8;
typedef __attribute__((ext_vector_type(4)))  int i32x4;

// ---------- helpers ----------
__device__ __forceinline__ unsigned short f2bf(float x) {
    unsigned int u = __float_as_uint(x);
    unsigned int r = u + 0x7fffu + ((u >> 16) & 1u);   // RNE
    return (unsigned short)(r >> 16);
}
__device__ __forceinline__ unsigned int fenc(float f) {   // monotonic f32->u32
    unsigned int u = __float_as_uint(f);
    return (u & 0x80000000u) ? ~u : (u | 0x80000000u);
}
__device__ __forceinline__ float fdec(unsigned int u) {
    return __uint_as_float((u & 0x80000000u) ? (u ^ 0x80000000u) : ~u);
}

// ---------- f32 -> bf16 conversion with zero padding (rows R->Rp, cols C->Cp) ----------
__global__ void k_cvt_bf16(const float* __restrict__ src, unsigned int* __restrict__ dst,
                           int R, int C, int Rp, int Cp)
{
    int i = blockIdx.x * blockDim.x + threadIdx.x;     // over Rp * Cp/2
    int half = Cp >> 1;
    if (i >= Rp * half) return;
    int r = i / half;
    int c = (i - r * half) << 1;
    float f0 = 0.f, f1 = 0.f;
    if (r < R) {
        const float* sp = src + (size_t)r * C + c;
        if (c     < C) f0 = sp[0];
        if (c + 1 < C) f1 = sp[1];
    }
    dst[i] = (unsigned int)f2bf(f0) | ((unsigned int)f2bf(f1) << 16);
}

// ---------- TDM: issue async tensor tile load (128 rows x 32 bf16) into LDS ----------
__device__ __forceinline__ void tdm_load_tile(const unsigned short* gbase, unsigned int ldsOff,
                                              int rowsRemaining, int lda)
{
    unsigned long long ga = (unsigned long long)(uintptr_t)gbase;
    unsigned int td0 = 32u;                       // K extent from tile start (>= tile_dim0)
    unsigned int td1 = (unsigned int)rowsRemaining;
    u32x4 g0;
    g0[0] = 1u;                                   // count=1, user mode, no gather
    g0[1] = ldsOff;                               // lds_addr (bytes)
    g0[2] = (unsigned int)ga;                     // global_addr[31:0]
    g0[3] = (unsigned int)((ga >> 32) & 0x01FFFFFFu) | 0x80000000u;  // addr[56:32] | type=2
    i32x8 g1;
    g1[0] = (int)(1u << 16);                      // data_size=1 -> 2 bytes
    g1[1] = (int)((td0 & 0xFFFFu) << 16);         // tensor_dim0[15:0]
    g1[2] = (int)(((td0 >> 16) & 0xFFFFu) | ((td1 & 0xFFFFu) << 16));
    g1[3] = (int)(((td1 >> 16) & 0xFFFFu) | (32u << 16));   // tile_dim0 = 32
    g1[4] = (int)128u;                            // tile_dim1 = 128, tile_dim2 = 0
    g1[5] = (int)(unsigned int)lda;               // tensor_dim0_stride low
    g1[6] = 0;                                    // stride0 hi | stride1 lo
    g1[7] = 0;
    i32x4 z4; z4[0] = 0; z4[1] = 0; z4[2] = 0; z4[3] = 0;
    i32x8 z8;
    #pragma unroll
    for (int q = 0; q < 8; ++q) z8[q] = 0;
    __builtin_amdgcn_tensor_load_to_lds(g0, g1, z4, z4, z8, 0);
}

// ---------- WMMA GEMM: C[M,N] = A[M,K]*B[K,N]; A,B bf16 row-major; K%32==0, N%64==0 ----------
// A tile staged by TDM (async tensor load, double buffered); B tile thread-staged transposed.
__global__ __launch_bounds__(256) void k_gemm_wmma(
    const unsigned short* __restrict__ A, const unsigned short* __restrict__ B,
    float* __restrict__ C, int M, int N, int K, int lda, int ldb, int ldc)
{
    __shared__ unsigned int sA[2][128 * 16];   // [buf][row][16 uints] row-major bf16 (16 KB)
    __shared__ unsigned int sB[2][64 * 16];    // [buf][n][16 uints] transposed bf16 (8 KB)

    const int tid  = threadIdx.x;
    const int lane = tid & 31;
    const int wave = tid >> 5;
    const int mBlock = blockIdx.y * 128;
    const int nBlock = blockIdx.x * 64;
    const int rowsRem = M - mBlock;

    v8f acc[4];
    #pragma unroll
    for (int j = 0; j < 4; ++j)
        #pragma unroll
        for (int i = 0; i < 8; ++i) acc[j][i] = 0.0f;

    const int nK = K >> 5;
    if (wave == 0)
        tdm_load_tile(A + (size_t)mBlock * lda, (unsigned int)(uintptr_t)&sA[0][0], rowsRem, lda);

    // B staging coords (uniform per thread across iterations)
    const int br = tid >> 3;            // k row within tile 0..31
    const int bn = (tid & 7) << 3;      // n offset 0..56 step 8

    for (int i = 0; i < nK; ++i) {
        const int p = i & 1;
        const int k0 = i << 5;

        // --- stage B tile transposed: 32x64 bf16, 8 elems/thread, no bounds needed ---
        {
            const unsigned short* bp = B + (size_t)(k0 + br) * ldb + nBlock + bn;
            uint4 v = *(const uint4*)bp;
            const unsigned short* sp = (const unsigned short*)&v;
            unsigned short* sBt = (unsigned short*)sB[p];
            #pragma unroll
            for (int q = 0; q < 8; ++q) sBt[(bn + q) * 32 + br] = sp[q];
            if (i + 1 < nK)
                __builtin_prefetch(bp + 32 * ldb, 0, 1);
        }

        if (wave == 0) __builtin_amdgcn_s_wait_tensorcnt(0);   // A(i) resident
        __syncthreads();

        // --- fragments + WMMA ---
        {
            const int hiA = (lane >= 16) ? 4 : 0;
            const int hiB = (lane >= 16) ? 4 : 0;
            const int arow = ((wave << 4) + (lane & 15)) << 4;
            union Frag { u32x4 u[2]; v16bf v; };
            Frag af;
            const unsigned int* pA = sA[p];
            const unsigned int* pB = sB[p];
            af.u[0] = *(const u32x4*)&pA[arow + hiA];
            af.u[1] = *(const u32x4*)&pA[arow + 8 + hiA];
            #pragma unroll
            for (int j = 0; j < 4; ++j) {
                Frag bf;
                int nrow = ((j << 4) + (lane & 15)) << 4;
                bf.u[0] = *(const u32x4*)&pB[nrow + hiB];
                bf.u[1] = *(const u32x4*)&pB[nrow + hiB + 8];
                acc[j] = __builtin_amdgcn_wmma_f32_16x16x32_bf16(
                    false, af.v, false, bf.v, (short)0, acc[j], false, false);
            }
        }
        __syncthreads();   // compute done everywhere before next buffers are written

        if (wave == 0 && i + 1 < nK)
            tdm_load_tile(A + (size_t)mBlock * lda + (k0 + 32),
                          (unsigned int)(uintptr_t)&sA[p ^ 1][0], rowsRem, lda);
    }

    // --- store C (f32 16x16 frag layout) ---
    const int rowBase = mBlock + (wave << 4) + ((lane >= 16) ? 8 : 0);
    const int col0 = nBlock + (lane & 15);
    #pragma unroll
    for (int j = 0; j < 4; ++j) {
        int col = col0 + j * 16;
        #pragma unroll
        for (int r = 0; r < 8; ++r) {
            int row = rowBase + r;
            if (row < M) C[(size_t)row * ldc + col] = acc[j][r];
        }
    }
}

// ---------- per-(row,head) dot: out[m][h] = sum_c H[m][h*256+c]*a[h][c] ----------
__global__ __launch_bounds__(256) void k_dot_heads(
    const float* __restrict__ H, const float* __restrict__ avec,
    float* __restrict__ out, int M)
{
    int gw = blockIdx.x * 8 + (threadIdx.x >> 5);
    int lane = threadIdx.x & 31;
    int m = gw >> 2, h = gw & 3;
    if (m >= M) return;
    const float* hp = H + (size_t)m * HC + h * HID;
    const float* ap = avec + h * HID;
    float s = 0.f;
    #pragma unroll
    for (int i = 0; i < 8; ++i) { int c = lane + 32 * i; s += hp[c] * ap[c]; }
    #pragma unroll
    for (int off = 16; off; off >>= 1) s += __shfl_xor(s, off, 32);
    if (lane == 0) out[m * 4 + h] = s;
}

__global__ void k_gather_rows(const float* __restrict__ T, const int* __restrict__ idx,
                              float* __restrict__ dst, int n)
{
    int i = blockIdx.x * blockDim.x + threadIdx.x;
    if (i >= n * (HC / 4)) return;
    int node = i / (HC / 4);
    int c4 = (i % (HC / 4)) * 4;
    *(float4*)(dst + (size_t)node * HC + c4) =
        *(const float4*)(T + (size_t)idx[node] * HC + c4);
}

__global__ void k_zero(unsigned int* p, long long n) {
    long long i = (long long)blockIdx.x * blockDim.x + threadIdx.x;
    if (i < n) p[i] = 0u;
}

__global__ void k_edge_logits(const int* __restrict__ src, const int* __restrict__ dst,
                              const int* __restrict__ et,
                              const float* __restrict__ als, const float* __restrict__ ald,
                              const float* __restrict__ ae,
                              float* __restrict__ logits, unsigned int* __restrict__ mmax, int E)
{
    int e = blockIdx.x * blockDim.x + threadIdx.x;
    if (e >= E) return;
    int s = src[e], d = dst[e], t = et[e];
    #pragma unroll
    for (int h = 0; h < HEADS; ++h) {
        float l = als[s * 4 + h] + ald[d * 4 + h] + ae[t * 4 + h];
        l = (l > 0.f) ? l : 0.2f * l;                 // leaky_relu
        logits[(size_t)e * 4 + h] = l;
        atomicMax(&mmax[d * 4 + h], fenc(l));
    }
}

__global__ void k_edge_exp(const int* __restrict__ dst,
                           float* __restrict__ ex, const unsigned int* __restrict__ mmax,
                           float* __restrict__ denom, int E)
{
    int e = blockIdx.x * blockDim.x + threadIdx.x;
    if (e >= E) return;
    int d = dst[e];
    #pragma unroll
    for (int h = 0; h < HEADS; ++h) {
        float v = __expf(ex[(size_t)e * 4 + h] - fdec(mmax[d * 4 + h]));
        ex[(size_t)e * 4 + h] = v;
        atomicAdd(&denom[d * 4 + h], v);
    }
}

__global__ __launch_bounds__(256) void k_edge_scatter(
    const int* __restrict__ src, const int* __restrict__ dst,
    const float* __restrict__ ex, const float* __restrict__ denom,
    const float* __restrict__ h, float* __restrict__ out)
{
    int e = blockIdx.x;
    int c = threadIdx.x;  // 0..255
    int s = src[e], d = dst[e];
    #pragma unroll
    for (int hd = 0; hd < HEADS; ++hd) {
        float a = ex[(size_t)e * 4 + hd] / (denom[d * 4 + hd] + 1e-16f);
        atomicAdd(&out[(size_t)d * HC + hd * HID + c],
                  a * h[(size_t)s * HC + hd * HID + c]);
    }
}

__global__ void k_bias_act(float* __restrict__ x, const float* __restrict__ b,
                           long long total, int elu)
{
    long long i = (long long)blockIdx.x * blockDim.x + threadIdx.x;
    if (i >= total) return;
    int c = (int)(i & (HC - 1));
    float v = x[i] + b[c];
    x[i] = elu ? ((v > 0.f) ? v : (__expf(v) - 1.f)) : v;
}

__global__ void k_pool_count(const int* __restrict__ batch, float* __restrict__ cnt, int N) {
    int i = blockIdx.x * blockDim.x + threadIdx.x;
    if (i < N) atomicAdd(&cnt[batch[i]], 1.0f);
}

__global__ void k_pool_sum(const float* __restrict__ x, const int* __restrict__ batch,
                           float* __restrict__ pooled, int N)
{
    long long i = (long long)blockIdx.x * blockDim.x + threadIdx.x;
    if (i >= (long long)N * HC) return;
    int n = (int)(i >> 10);
    int c = (int)(i & (HC - 1));
    atomicAdd(&pooled[(size_t)batch[n] * HC + c], x[i]);
}

__global__ __launch_bounds__(256) void k_head(
    const float* __restrict__ pooled, const float* __restrict__ cnt,
    const float* __restrict__ Wp, const float* __restrict__ bp,
    const float* __restrict__ lng, const float* __restrict__ lnb,
    float* __restrict__ out)
{
    __shared__ float row[HC];
    __shared__ float red[HID];
    int g = blockIdx.x, t = threadIdx.x;
    float c = fmaxf(cnt[g], 1.0f);
    for (int i = t; i < HC; i += HID) row[i] = pooled[(size_t)g * HC + i] / c;
    __syncthreads();
    float y = bp[t];
    for (int k = 0; k < HC; ++k) y += row[k] * Wp[(size_t)k * HID + t];
    red[t] = y; __syncthreads();
    for (int s = HID / 2; s; s >>= 1) { if (t < s) red[t] += red[t + s]; __syncthreads(); }
    float mu = red[0] / (float)HID;
    __syncthreads();
    float d0 = y - mu;
    red[t] = d0 * d0; __syncthreads();
    for (int s = HID / 2; s; s >>= 1) { if (t < s) red[t] += red[t + s]; __syncthreads(); }
    float var = red[0] / (float)HID;
    float r = d0 * rsqrtf(var + 1e-5f) * lng[t] + lnb[t];
    out[(size_t)g * HID + t] = fmaxf(r, 0.0f);
}

// =====================================================================
extern "C" void kernel_launch(void* const* d_in, const int* in_sizes, int n_in,
                              void* d_out, int out_size, void* d_ws, size_t ws_size,
                              hipStream_t stream) {
    (void)in_sizes; (void)n_in; (void)out_size; (void)ws_size;

    const int*   x_idx    = (const int*)d_in[0];
    const int*   eidx     = (const int*)d_in[1];
    const int*   src      = eidx;
    const int*   dst      = eidx + N_EDGES;
    const int*   etype    = (const int*)d_in[2];
    const int*   batch    = (const int*)d_in[3];
    const float* node_emb = (const float*)d_in[4];
    const float* edge_emb = (const float*)d_in[5];
    const float* W1       = (const float*)d_in[6];
    const float* a_src1   = (const float*)d_in[7];
    const float* a_dst1   = (const float*)d_in[8];
    const float* a_edge1  = (const float*)d_in[9];
    const float* We1      = (const float*)d_in[10];
    const float* b1       = (const float*)d_in[11];
    const float* W2       = (const float*)d_in[12];
    const float* a_src2   = (const float*)d_in[13];
    const float* a_dst2   = (const float*)d_in[14];
    const float* a_edge2  = (const float*)d_in[15];
    const float* We2      = (const float*)d_in[16];
    const float* b2       = (const float*)d_in[17];
    const float* projW    = (const float*)d_in[18];
    const float* projb    = (const float*)d_in[19];
    const float* lng      = (const float*)d_in[20];
    const float* lnb      = (const float*)d_in[21];

    // --- workspace carve ---
    char* w = (char*)d_ws;
    size_t off = 0;
    auto carve = [&](size_t bytes) -> void* {
        void* p = w + off;
        off = (off + bytes + 255) & ~(size_t)255;
        return p;
    };
    float* T1    = (float*)carve((size_t)N_EMB   * HC * 4);
    float* h     = (float*)carve((size_t)N_NODES * HC * 4);
    float* xo    = (float*)carve((size_t)N_NODES * HC * 4);
    float* out2  = (float*)carve((size_t)N_NODES * HC * 4);
    float* ET1   = (float*)carve((size_t)N_TYPES * HC * 4);
    float* ET2   = (float*)carve((size_t)N_TYPES * HC * 4);
    float* ae1   = (float*)carve((size_t)N_TYPES * 4 * 4);
    float* ae2   = (float*)carve((size_t)N_TYPES * 4 * 4);
    float* als   = (float*)carve((size_t)N_NODES * 4 * 4);
    float* ald   = (float*)carve((size_t)N_NODES * 4 * 4);
    unsigned int* mmax = (unsigned int*)carve((size_t)N_NODES * 4 * 4);
    float* denom = (float*)carve((size_t)N_NODES * 4 * 4);
    float* exbuf = (float*)carve((size_t)N_EDGES * 4 * 4);
    float* pooled= (float*)carve((size_t)N_GRAPHS * HC * 4);
    float* cnt   = (float*)carve((size_t)N_GRAPHS * 4);
    // bf16 operand copies for WMMA/TDM GEMMs
    unsigned short* ne_bf = (unsigned short*)carve((size_t)N_EMB * FEATP * 2);
    unsigned short* W1_bf = (unsigned short*)carve((size_t)FEATP * HC * 2);
    unsigned short* ee_bf = (unsigned short*)carve((size_t)N_TYPES * HC * 2);
    unsigned short* We1_bf= (unsigned short*)carve((size_t)HC * HC * 2);
    unsigned short* We2_bf= (unsigned short*)carve((size_t)HC * HC * 2);
    unsigned short* W2_bf = (unsigned short*)carve((size_t)HC * HC * 2);
    unsigned short* x2_bf = (unsigned short*)carve((size_t)N_NODES * HC * 2);

    const long long nodeElems = (long long)N_NODES * HC;
    const int eBlocks = (N_EDGES + 255) / 256;
    const int dotBlocksN = (N_NODES * 4 + 7) / 8;
    const int dotBlocksT = (N_TYPES * 4 + 7) / 8;
    auto zgrid = [](long long n) { return (int)((n + 255) / 256); };
    auto cvtGrid = [](int Rp, int Cp) { return (Rp * (Cp / 2) + 255) / 256; };

    // ---- phase A: bf16 conversions + WMMA table GEMMs ----
    k_cvt_bf16<<<cvtGrid(N_EMB, FEATP), 256, 0, stream>>>(node_emb, (unsigned int*)ne_bf, N_EMB, FEAT, N_EMB, FEATP);
    k_cvt_bf16<<<cvtGrid(FEATP, HC),   256, 0, stream>>>(W1,       (unsigned int*)W1_bf, FEAT, HC, FEATP, HC);
    k_cvt_bf16<<<cvtGrid(N_TYPES, HC), 256, 0, stream>>>(edge_emb, (unsigned int*)ee_bf, N_TYPES, HC, N_TYPES, HC);
    k_cvt_bf16<<<cvtGrid(HC, HC),      256, 0, stream>>>(We1,      (unsigned int*)We1_bf, HC, HC, HC, HC);
    k_cvt_bf16<<<cvtGrid(HC, HC),      256, 0, stream>>>(We2,      (unsigned int*)We2_bf, HC, HC, HC, HC);
    k_cvt_bf16<<<cvtGrid(HC, HC),      256, 0, stream>>>(W2,       (unsigned int*)W2_bf, HC, HC, HC, HC);

    k_gemm_wmma<<<dim3(HC / 64, (N_EMB + 127) / 128), 256, 0, stream>>>(
        ne_bf, W1_bf, T1, N_EMB, HC, FEATP, FEATP, HC, HC);
    k_gemm_wmma<<<dim3(HC / 64, (N_TYPES + 127) / 128), 256, 0, stream>>>(
        ee_bf, We1_bf, ET1, N_TYPES, HC, HC, HC, HC, HC);
    k_gemm_wmma<<<dim3(HC / 64, (N_TYPES + 127) / 128), 256, 0, stream>>>(
        ee_bf, We2_bf, ET2, N_TYPES, HC, HC, HC, HC, HC);
    k_dot_heads<<<dotBlocksT, 256, 0, stream>>>(ET1, a_edge1, ae1, N_TYPES);
    k_dot_heads<<<dotBlocksT, 256, 0, stream>>>(ET2, a_edge2, ae2, N_TYPES);

    // ---- layer 1 ----
    k_gather_rows<<<(N_NODES * (HC / 4) + 255) / 256, 256, 0, stream>>>(T1, x_idx, h, N_NODES);
    k_dot_heads<<<dotBlocksN, 256, 0, stream>>>(h, a_src1, als, N_NODES);
    k_dot_heads<<<dotBlocksN, 256, 0, stream>>>(h, a_dst1, ald, N_NODES);
    k_zero<<<zgrid(N_NODES * 4), 256, 0, stream>>>(mmax, (long long)N_NODES * 4);
    k_zero<<<zgrid(N_NODES * 4), 256, 0, stream>>>((unsigned int*)denom, (long long)N_NODES * 4);
    k_zero<<<zgrid(nodeElems), 256, 0, stream>>>((unsigned int*)xo, nodeElems);
    k_edge_logits<<<eBlocks, 256, 0, stream>>>(src, dst, etype, als, ald, ae1, exbuf, mmax, N_EDGES);
    k_edge_exp<<<eBlocks, 256, 0, stream>>>(dst, exbuf, mmax, denom, N_EDGES);
    k_edge_scatter<<<N_EDGES, 256, 0, stream>>>(src, dst, exbuf, denom, h, xo);
    k_bias_act<<<zgrid(nodeElems), 256, 0, stream>>>(xo, b1, nodeElems, 1);

    // ---- layer 2 ----
    k_cvt_bf16<<<cvtGrid(N_NODES, HC), 256, 0, stream>>>(xo, (unsigned int*)x2_bf, N_NODES, HC, N_NODES, HC);
    k_gemm_wmma<<<dim3(HC / 64, (N_NODES + 127) / 128), 256, 0, stream>>>(
        x2_bf, W2_bf, h, N_NODES, HC, HC, HC, HC, HC);
    k_dot_heads<<<dotBlocksN, 256, 0, stream>>>(h, a_src2, als, N_NODES);
    k_dot_heads<<<dotBlocksN, 256, 0, stream>>>(h, a_dst2, ald, N_NODES);
    k_zero<<<zgrid(N_NODES * 4), 256, 0, stream>>>(mmax, (long long)N_NODES * 4);
    k_zero<<<zgrid(N_NODES * 4), 256, 0, stream>>>((unsigned int*)denom, (long long)N_NODES * 4);
    k_zero<<<zgrid(nodeElems), 256, 0, stream>>>((unsigned int*)out2, nodeElems);
    k_edge_logits<<<eBlocks, 256, 0, stream>>>(src, dst, etype, als, ald, ae2, exbuf, mmax, N_EDGES);
    k_edge_exp<<<eBlocks, 256, 0, stream>>>(dst, exbuf, mmax, denom, N_EDGES);
    k_edge_scatter<<<N_EDGES, 256, 0, stream>>>(src, dst, exbuf, denom, h, out2);
    k_bias_act<<<zgrid(nodeElems), 256, 0, stream>>>(out2, b2, nodeElems, 0);

    // ---- pool + head ----
    k_zero<<<zgrid(N_GRAPHS * HC), 256, 0, stream>>>((unsigned int*)pooled, (long long)N_GRAPHS * HC);
    k_zero<<<1, 256, 0, stream>>>((unsigned int*)cnt, N_GRAPHS);
    k_pool_count<<<(N_NODES + 255) / 256, 256, 0, stream>>>(batch, cnt, N_NODES);
    k_pool_sum<<<zgrid(nodeElems), 256, 0, stream>>>(out2, batch, pooled, N_NODES);
    k_head<<<N_GRAPHS, HID, 0, stream>>>(pooled, cnt, projW, projb, lng, lnb, (float*)d_out);
}
